// Context_Adaptive_Gated_Attention_77292231459530
// MI455X (gfx1250) — matrix operand order repaired and checked by reading, not compiled, not run on hardware
//
#include <hip/hip_runtime.h>
#include <hip/hip_bf16.h>

typedef __attribute__((ext_vector_type(16))) _Float16 v16h;
typedef __attribute__((ext_vector_type(8)))  _Float16 v8h;
typedef __attribute__((ext_vector_type(4)))  _Float16 v4h;
typedef __attribute__((ext_vector_type(8)))  float    v8f;

#define DIM    384
#define QKVC   1152
#define NPIX   9216
#define KDIM   384
#define HEADS  8
#define HD     48

// ---------------------------------------------------------------------------
// WMMA + async helpers
// ---------------------------------------------------------------------------
__device__ __forceinline__ v8f wmma_f16(v16h a, v16h b, v8f c) {
  return __builtin_amdgcn_wmma_f32_16x16x32_f16(
      false, a, false, b, (short)0, c, false, false);
}

// CDNA5 async global->LDS copy (ASYNCcnt path), 16 bytes per lane.
__device__ __forceinline__ void async_copy_b128(unsigned int lds_off,
                                                const void* gaddr) {
  asm volatile("global_load_async_to_lds_b128 %0, %1, off"
               :: "v"(lds_off), "v"(gaddr) : "memory");
}
__device__ __forceinline__ void wait_asynccnt0() {
  asm volatile("s_wait_asynccnt 0x0" ::: "memory");
}

// A fragment: 16x32 (MxK). LDS row-major [m][k], stride lda halves (even).
__device__ __forceinline__ v16h load_a_frag(const _Float16* sA, int m0, int lda,
                                            int k0, int lane) {
  const int m   = m0 + (lane & 15);
  const int klo = (lane & 16) ? 16 : 0;
  const _Float16* row = sA + m * lda + k0 + klo;
  v16h a;
#pragma unroll
  for (int r = 0; r < 8; ++r) {
    const int k = 2 * r;
    a[2 * r]     = row[k];
    a[2 * r + 1] = row[k + 1];
  }
  return a;
}

// B fragment: 32x16 (KxN). LDS stored TRANSPOSED [n][k], stride ldb halves.
__device__ __forceinline__ v16h load_b_frag(const _Float16* sBt, int n0, int ldb,
                                            int k0, int lane) {
  const int n  = n0 + (lane & 15);
  const int kb = k0 + ((lane & 16) ? 16 : 0);
  const _Float16* p = sBt + n * ldb + kb;
  v16h b;
#pragma unroll
  for (int h = 0; h < 16; ++h) b[h] = p[h];
  return b;
}

// ---------------------------------------------------------------------------
// 1. Context MLPs: temperature per (b,head), sigmoid gate per (b,channel)
// ---------------------------------------------------------------------------
__global__ __launch_bounds__(384) void context_kernel(
    const float* __restrict__ ctx, const float* __restrict__ ta_w1,
    const float* __restrict__ ta_b1, const float* __restrict__ ta_w2,
    const float* __restrict__ ta_b2, const float* __restrict__ logbase,
    const float* __restrict__ gate_w, const float* __restrict__ gate_b,
    float* __restrict__ temp, float* __restrict__ gateo) {
  const int b = blockIdx.x;
  const int t = threadIdx.x;
  __shared__ float sc[512];
  __shared__ float hid[96];
  for (int i = t; i < 512; i += 384) sc[i] = ctx[b * 512 + i];
  __syncthreads();
  if (t < 96) {
    float a = ta_b1[t];
    for (int s = 0; s < 512; ++s) a += sc[s] * ta_w1[t * 512 + s];
    hid[t] = fmaxf(a, 0.0f);
  }
  __syncthreads();
  if (t < 8) {
    float a = ta_b2[t];
    for (int i = 0; i < 96; ++i) a += hid[i] * ta_w2[t * 96 + i];
    temp[b * 8 + t] = __expf(logbase[t] + a);
  }
  {
    float a = gate_b[t];
    for (int s = 0; s < 512; ++s) a += sc[s] * gate_w[t * 512 + s];
    gateo[b * 384 + t] = 1.0f / (1.0f + __expf(-a));
  }
}

// ---------------------------------------------------------------------------
// 2. qkv 1x1 conv WMMA GEMM, M=pixels / N=channels so the D-tile maps to
//    8 consecutive pixels per lane -> single b128 fp16 store per tile.
// ---------------------------------------------------------------------------
__global__ __launch_bounds__(128) void gemm_qkv_kernel(
    const float* __restrict__ X,     // [8][384][9216]
    const float* __restrict__ W,     // [1152][384]
    const float* __restrict__ bias,  // [1152]
    _Float16* __restrict__ Out)      // [8][1152][9216] fp16
{
  __shared__ _Float16 sXt[64 * 40];  // [pix][k], transposed staging
  __shared__ _Float16 sW[64 * 32];   // [co][k], direct row copy
  const int tid  = threadIdx.x;
  const int lane = tid & 31;
  const int wave = tid >> 5;
  const int pix0 = blockIdx.x * 64;
  const int co0  = blockIdx.y * 64;
  const int b    = blockIdx.z;
  const float* Xb = X + (size_t)b * KDIM * NPIX;

  v8f acc[4];
#pragma unroll
  for (int nt = 0; nt < 4; ++nt) {
    const float bv = bias[co0 + nt * 16 + (lane & 15)];
#pragma unroll
    for (int r = 0; r < 8; ++r) acc[nt][r] = bv;
  }

  for (int k0 = 0; k0 < KDIM; k0 += 32) {
    __syncthreads();
    // stage X^T: float4 along pixels, scatter 4 halves (transpose)
#pragma unroll
    for (int it = 0; it < 4; ++it) {
      const int q = tid + 128 * it;          // 0..511
      const int j = q >> 4;                  // k
      const int p4 = (q & 15) * 4;           // pixel group
      const float4 x4 = *(const float4*)&Xb[(size_t)(k0 + j) * NPIX + pix0 + p4];
      sXt[(p4 + 0) * 40 + j] = (_Float16)x4.x;
      sXt[(p4 + 1) * 40 + j] = (_Float16)x4.y;
      sXt[(p4 + 2) * 40 + j] = (_Float16)x4.z;
      sXt[(p4 + 3) * 40 + j] = (_Float16)x4.w;
    }
    // stage W rows: float4 loads -> v4h contiguous stores
#pragma unroll
    for (int it = 0; it < 4; ++it) {
      const int q = tid + 128 * it;
      const int i = q >> 3, j4 = (q & 7) * 4;
      const float4 w4 = *(const float4*)&W[(size_t)(co0 + i) * KDIM + k0 + j4];
      v4h hv;
      hv[0] = (_Float16)w4.x; hv[1] = (_Float16)w4.y;
      hv[2] = (_Float16)w4.z; hv[3] = (_Float16)w4.w;
      *(v4h*)&sW[i * 32 + j4] = hv;
    }
    __syncthreads();
    const v16h a = load_a_frag(sXt, wave * 16, 40, 0, lane);
#pragma unroll
    for (int nt = 0; nt < 4; ++nt)
      acc[nt] = wmma_f16(a, load_b_frag(sW, nt * 16, 32, 0, lane), acc[nt]);
  }

  const int pixbase = pix0 + wave * 16 + 8 * (lane >> 4);
  _Float16* Ob = Out + (size_t)b * QKVC * NPIX;
#pragma unroll
  for (int nt = 0; nt < 4; ++nt) {
    const int co = co0 + nt * 16 + (lane & 15);
    v8h pk;
#pragma unroll
    for (int r = 0; r < 8; ++r) pk[r] = (_Float16)acc[nt][r];
    *(v8h*)&Ob[(size_t)co * NPIX + pixbase] = pk;
  }
}

// ---------------------------------------------------------------------------
// 3. Depthwise 3x3, SAME. 8 pixels/thread, v8h row loads, fp32 accum.
// ---------------------------------------------------------------------------
__global__ __launch_bounds__(128) void dwconv_kernel(
    const _Float16* __restrict__ In, const float* __restrict__ Wdw,
    const float* __restrict__ bias, _Float16* __restrict__ Outp) {
  const int c = blockIdx.y;
  const int b = blockIdx.z;
  const int g = blockIdx.x * 128 + threadIdx.x;  // 0..1151
  const int y = g / 12, x8 = (g % 12) * 8;
  const _Float16* p = In + ((size_t)b * QKVC + c) * NPIX;
  float w[9];
#pragma unroll
  for (int i = 0; i < 9; ++i) w[i] = Wdw[c * 9 + i];
  float acc[8];
#pragma unroll
  for (int u = 0; u < 8; ++u) acc[u] = bias[c];
#pragma unroll
  for (int ky = 0; ky < 3; ++ky) {
    const int yy = y + ky - 1;
    if (yy < 0 || yy >= 96) continue;
    const _Float16* rowp = p + yy * 96 + x8;
    const v8h ctr = *(const v8h*)rowp;
    float e[10];
    e[0] = (x8 > 0) ? (float)rowp[-1] : 0.0f;
    e[9] = (x8 + 8 < 96) ? (float)rowp[8] : 0.0f;
#pragma unroll
    for (int u = 0; u < 8; ++u) e[u + 1] = (float)ctr[u];
#pragma unroll
    for (int kx = 0; kx < 3; ++kx) {
      const float wv = w[ky * 3 + kx];
#pragma unroll
      for (int u = 0; u < 8; ++u) acc[u] += wv * e[u + kx];
    }
  }
  v8h o;
#pragma unroll
  for (int u = 0; u < 8; ++u) o[u] = (_Float16)acc[u];
  *(v8h*)(Outp + ((size_t)b * QKVC + c) * NPIX + y * 96 + x8) = o;
}

// ---------------------------------------------------------------------------
// 4. Row L2 inverse norms for q and k.
// ---------------------------------------------------------------------------
__global__ __launch_bounds__(256) void norm_kernel(
    const _Float16* __restrict__ D, float* __restrict__ invq,
    float* __restrict__ invk) {
  const int row = blockIdx.x;        // b*384 + head*48 + c
  const int b = row / 384;
  const int hc = row % 384;
  const _Float16* q = D + ((size_t)b * QKVC + hc) * NPIX;
  const _Float16* k = q + (size_t)DIM * NPIX;
  float sq = 0.f, sk = 0.f;
  for (int i = threadIdx.x; i < NPIX / 8; i += 256) {
    const v8h qa = *(const v8h*)&q[i * 8];
    const v8h ka = *(const v8h*)&k[i * 8];
#pragma unroll
    for (int u = 0; u < 8; ++u) {
      const float a = (float)qa[u]; sq += a * a;
      const float c2 = (float)ka[u]; sk += c2 * c2;
    }
  }
  __shared__ float rq[256], rk[256];
  rq[threadIdx.x] = sq; rk[threadIdx.x] = sk;
  __syncthreads();
  for (int s = 128; s > 0; s >>= 1) {
    if (threadIdx.x < s) {
      rq[threadIdx.x] += rq[threadIdx.x + s];
      rk[threadIdx.x] += rk[threadIdx.x + s];
    }
    __syncthreads();
  }
  if (threadIdx.x == 0) {
    invq[row] = 1.0f / fmaxf(sqrtf(rq[0]), 1e-12f);
    invk[row] = 1.0f / fmaxf(sqrtf(rk[0]), 1e-12f);
  }
}

// ---------------------------------------------------------------------------
// 5. Per-(b,head) attention. Output TRANSPOSED: Yt[b][n][c] fp16.
// ---------------------------------------------------------------------------
__global__ __launch_bounds__(96) void attn_kernel(
    const _Float16* __restrict__ D, const float* __restrict__ invq,
    const float* __restrict__ invk, const float* __restrict__ temp,
    const float* __restrict__ gate, _Float16* __restrict__ Yt) {
  const int bh = blockIdx.x;
  const int b = bh >> 3, h = bh & 7;
  const int tid = threadIdx.x, lane = tid & 31, wave = tid >> 5;
  const _Float16* Q  = D + ((size_t)b * QKVC + h * HD) * NPIX;
  const _Float16* Kp = Q + (size_t)DIM * NPIX;
  const _Float16* V  = Q + (size_t)(2 * DIM) * NPIX;
  const float* iq = invq + (b * HEADS + h) * HD;
  const float* ik = invk + (b * HEADS + h) * HD;

  __shared__ _Float16 sQ[48 * 40];   // [c][n] stride 40
  __shared__ _Float16 sK[48 * 40];   // [d][n]
  __shared__ float    sS[48 * 48];
  __shared__ _Float16 sSh[48 * 64];  // softmaxed S, K padded to 64
  __shared__ _Float16 sVt[32 * 64];  // [n][d], d padded to 64

  v8f acc[3];
#pragma unroll
  for (int t2 = 0; t2 < 3; ++t2)
#pragma unroll
    for (int r = 0; r < 8; ++r) acc[t2][r] = 0.0f;

  // ---- Gram: S[c][d] = sum_n qhat[c][n] khat[d][n] ----
  for (int n0 = 0; n0 < NPIX; n0 += 32) {
    __syncthreads();
#pragma unroll
    for (int it = 0; it < 2; ++it) {
      const int q8 = tid + 96 * it;            // 0..191
      const int c = q8 >> 2, j8 = (q8 & 3) * 8;
      const v8h qa = *(const v8h*)&Q[(size_t)c * NPIX + n0 + j8];
      const v8h ka = *(const v8h*)&Kp[(size_t)c * NPIX + n0 + j8];
      const float qs = iq[c], ks = ik[c];
      v8h qo, ko;
#pragma unroll
      for (int u = 0; u < 8; ++u) {
        qo[u] = (_Float16)((float)qa[u] * qs);
        ko[u] = (_Float16)((float)ka[u] * ks);
      }
      *(v8h*)&sQ[c * 40 + j8] = qo;
      *(v8h*)&sK[c * 40 + j8] = ko;
    }
    __syncthreads();
    const v16h a = load_a_frag(sQ, wave * 16, 40, 0, lane);
#pragma unroll
    for (int dt = 0; dt < 3; ++dt)
      acc[dt] = wmma_f16(a, load_b_frag(sK, dt * 16, 40, 0, lane), acc[dt]);
  }

  __syncthreads();
  const int mbase = wave * 16 + 8 * (lane >> 4);
  const int nc = lane & 15;
#pragma unroll
  for (int dt = 0; dt < 3; ++dt)
#pragma unroll
    for (int r = 0; r < 8; ++r)
      sS[(mbase + r) * 48 + dt * 16 + nc] = acc[dt][r];
  // zero-pad persistent parts once: sVt columns d=48..63
  for (int e = tid; e < 512; e += 96) {
    const int j = e >> 4, d = 48 + (e & 15);
    sVt[j * 64 + d] = (_Float16)0.0f;
  }
  __syncthreads();

  // ---- temperature + row softmax -> fp16 (K padded to 64) ----
  if (tid < 48) {
    const float tmp = temp[b * HEADS + h];
    float mx = -3.0e30f;
    for (int d = 0; d < 48; ++d) {
      const float v2 = sS[tid * 48 + d] * tmp;
      sS[tid * 48 + d] = v2;
      mx = fmaxf(mx, v2);
    }
    float sum = 0.f;
    for (int d = 0; d < 48; ++d) {
      const float e2 = __expf(sS[tid * 48 + d] - mx);
      sS[tid * 48 + d] = e2;
      sum += e2;
    }
    const float inv = 1.0f / sum;
    for (int d = 0; d < 48; ++d) sSh[tid * 64 + d] = (_Float16)(sS[tid * 48 + d] * inv);
    for (int d = 48; d < 64; ++d) sSh[tid * 64 + d] = (_Float16)0.0f;
  }
  __syncthreads();

  float gv[8];
#pragma unroll
  for (int r = 0; r < 8; ++r) gv[r] = gate[b * DIM + h * HD + mbase + r];

  // ---- O = S V, gate fused, transposed b128 stores ----
  for (int n0 = 0; n0 < NPIX; n0 += 32) {
    __syncthreads();
#pragma unroll
    for (int it = 0; it < 2; ++it) {
      const int q8 = tid + 96 * it;            // 0..191
      const int d = q8 >> 2, j8 = (q8 & 3) * 8;
      const v8h va = *(const v8h*)&V[(size_t)d * NPIX + n0 + j8];
#pragma unroll
      for (int u = 0; u < 8; ++u) sVt[(j8 + u) * 64 + d] = va[u];
    }
    __syncthreads();
#pragma unroll
    for (int nt = 0; nt < 2; ++nt) {
      v8f o;
#pragma unroll
      for (int r = 0; r < 8; ++r) o[r] = 0.0f;
#pragma unroll
      for (int kc = 0; kc < 2; ++kc) {
        const v16h a2 = load_a_frag(sSh, wave * 16, 64, kc * 32, lane);
        const v16h b2 = load_b_frag(sVt, nt * 16, 64, kc * 32, lane);
        o = wmma_f16(a2, b2, o);
      }
      const int n = n0 + nt * 16 + nc;
      v8h pack;
#pragma unroll
      for (int r = 0; r < 8; ++r) pack[r] = (_Float16)(o[r] * gv[r]);
      *(v8h*)&Yt[((size_t)b * NPIX + n) * DIM + h * HD + mbase] = pack;
    }
  }
}

// ---------------------------------------------------------------------------
// 6. proj WMMA GEMM, M=pixels / N=channels. A tile = contiguous Yt rows ->
//    GLOBAL_LOAD_ASYNC_TO_LDS_B128; fp32 out as two float4 per lane.
// ---------------------------------------------------------------------------
__global__ __launch_bounds__(128) void gemm_proj_kernel(
    const _Float16* __restrict__ Yt,   // [8][9216][384] fp16
    const float* __restrict__ W,       // [384][384]
    const float* __restrict__ bias,    // [384]
    float* __restrict__ Out)           // [8][384][9216] fp32
{
  __shared__ _Float16 sYt[64 * 32];    // [pix][k], async-filled (contiguous)
  __shared__ _Float16 sW[64 * 40];     // [co][k]
  const int tid  = threadIdx.x;
  const int lane = tid & 31;
  const int wave = tid >> 5;
  const int pix0 = blockIdx.x * 64;
  const int co0  = blockIdx.y * 64;
  const int b    = blockIdx.z;
  const _Float16* Ytb = Yt + (size_t)b * NPIX * KDIM;
  const unsigned int ldsA = (unsigned int)(uintptr_t)(&sYt[0]);

  v8f acc[4];
#pragma unroll
  for (int nt = 0; nt < 4; ++nt) {
    const float bv = bias[co0 + nt * 16 + (lane & 15)];
#pragma unroll
    for (int r = 0; r < 8; ++r) acc[nt][r] = bv;
  }

  for (int k0 = 0; k0 < KDIM; k0 += 32) {
    __syncthreads();
    // async stage A: 64 Yt rows x 64 bytes, 16B per lane-copy
#pragma unroll
    for (int it = 0; it < 2; ++it) {
      const int o = (tid + 128 * it) * 16;     // 0..4095
      const int row = o >> 6, colb = o & 63;
      const char* g = (const char*)Ytb +
                      (((size_t)(pix0 + row) * KDIM + k0) * 2 + colb);
      async_copy_b128(ldsA + (unsigned int)(row * 64 + colb), g);
    }
    // stage W rows (fp32 -> fp16)
#pragma unroll
    for (int it = 0; it < 4; ++it) {
      const int q = tid + 128 * it;
      const int i = q >> 3, j4 = (q & 7) * 4;
      const float4 w4 = *(const float4*)&W[(size_t)(co0 + i) * KDIM + k0 + j4];
      v4h hv;
      hv[0] = (_Float16)w4.x; hv[1] = (_Float16)w4.y;
      hv[2] = (_Float16)w4.z; hv[3] = (_Float16)w4.w;
      *(v4h*)&sW[i * 40 + j4] = hv;
    }
    wait_asynccnt0();
    __syncthreads();
    const v16h a = load_a_frag(sYt, wave * 16, 32, 0, lane);
#pragma unroll
    for (int nt = 0; nt < 4; ++nt)
      acc[nt] = wmma_f16(a, load_b_frag(sW, nt * 16, 40, 0, lane), acc[nt]);
  }

  const int pixbase = pix0 + wave * 16 + 8 * (lane >> 4);
  float* Ob = Out + (size_t)b * DIM * NPIX;
#pragma unroll
  for (int nt = 0; nt < 4; ++nt) {
    const int co = co0 + nt * 16 + (lane & 15);
    float4 lo, hi;
    lo.x = acc[nt][0]; lo.y = acc[nt][1]; lo.z = acc[nt][2]; lo.w = acc[nt][3];
    hi.x = acc[nt][4]; hi.y = acc[nt][5]; hi.z = acc[nt][6]; hi.w = acc[nt][7];
    *(float4*)&Ob[(size_t)co * NPIX + pixbase] = lo;
    *(float4*)&Ob[(size_t)co * NPIX + pixbase + 4] = hi;
  }
}

// ---------------------------------------------------------------------------
extern "C" void kernel_launch(void* const* d_in, const int* in_sizes, int n_in,
                              void* d_out, int out_size, void* d_ws, size_t ws_size,
                              hipStream_t stream) {
  const float* x       = (const float*)d_in[0];
  const float* ctx     = (const float*)d_in[1];
  const float* qkv_w   = (const float*)d_in[2];
  const float* qkv_b   = (const float*)d_in[3];
  const float* dw_w    = (const float*)d_in[4];
  const float* dw_b    = (const float*)d_in[5];
  const float* proj_w  = (const float*)d_in[6];
  const float* proj_b  = (const float*)d_in[7];
  const float* logbase = (const float*)d_in[8];
  const float* ta_w1   = (const float*)d_in[9];
  const float* ta_b1   = (const float*)d_in[10];
  const float* ta_w2   = (const float*)d_in[11];
  const float* ta_b2   = (const float*)d_in[12];
  const float* gate_w  = (const float*)d_in[13];
  const float* gate_b  = (const float*)d_in[14];
  float* out = (float*)d_out;

  char* ws = (char*)d_ws;
  const size_t SZ_QKV = (size_t)8 * QKVC * NPIX * sizeof(_Float16);
  const size_t SZ_Y   = (size_t)8 * DIM  * NPIX * sizeof(_Float16);
  _Float16* qkv16 = (_Float16*)ws;
  _Float16* dw16  = (_Float16*)(ws + SZ_QKV);
  _Float16* y16t  = (_Float16*)(ws + 2 * SZ_QKV);     // [b][n][c]
  float* temp = (float*)(ws + 2 * SZ_QKV + SZ_Y);
  float* gate = temp + 64;
  float* invq = gate + 8 * DIM;
  float* invk = invq + 8 * HEADS * HD;

  context_kernel<<<8, 384, 0, stream>>>(ctx, ta_w1, ta_b1, ta_w2, ta_b2,
                                        logbase, gate_w, gate_b, temp, gate);
  gemm_qkv_kernel<<<dim3(NPIX / 64, QKVC / 64, 8), 128, 0, stream>>>(
      x, qkv_w, qkv_b, qkv16);
  dwconv_kernel<<<dim3(9, QKVC, 8), 128, 0, stream>>>(qkv16, dw_w, dw_b, dw16);
  norm_kernel<<<8 * HEADS * HD, 256, 0, stream>>>(dw16, invq, invk);
  attn_kernel<<<8 * HEADS, 96, 0, stream>>>(dw16, invq, invk, temp, gate, y16t);
  gemm_proj_kernel<<<dim3(NPIX / 64, DIM / 64, 8), 128, 0, stream>>>(
      y16t, proj_w, proj_b, out);
}
